// SupernodePooling_23063974379705
// MI455X (gfx1250) — matrix operand order repaired and checked by reading, not compiled, not run on hardware
//
#include <hip/hip_runtime.h>
#include <hip/hip_bf16.h>
#include <math.h>

// ---------------------------------------------------------------------------
// SupernodePooling on MI455X (gfx1250, wave32, WMMA 16x16x32 f16->f32)
//
// B=2, N=4096, IN_DIM=16, HID=256, MAX_SUP=1024, RADIUS=0.12
// Pipeline:
//   1) feat_kernel : x[n] = feat@w_in + b_in + sincos(pos)   -> f16 copy xh
//   2) wcvt_kernel : f16 copies of W1_top (src half) and W2
//   3) edge_kernel : per-supernode radius neighbor list (cap 64)
//   4) dvec_kernel : dvec[s] = x[s]@W1_bot + b1   (per-supernode constant)
//   5) mlp_pool    : per supernode, tiles of 16 edges:
//        h1 = gelu(Xsrc(16x256) @ W1_top + dvec)   [16 WMMA tiles x 8 ksteps]
//        o  = h1 @ W2                               [16 WMMA tiles x 8 ksteps]
//        accv += rowsum(o) over valid edges; out = accv/cnt + b2*(cnt>0)
// ---------------------------------------------------------------------------

#define B_       2
#define N_       4096
#define NDIM_    3
#define IN_DIM_  16
#define HID_     256
#define MAX_SUP_ 1024
#define CAP_     64
#define R2_      (0.12f * 0.12f)

typedef __attribute__((ext_vector_type(16))) _Float16 v16h;
typedef __attribute__((ext_vector_type(8)))  _Float16 v8h;
typedef __attribute__((ext_vector_type(8)))  float    v8f;

__device__ __forceinline__ float gelu_tanh(float x) {
    // flax approximate=True gelu
    float t = tanhf(0.7978845608028654f * (x + 0.044715f * x * x * x));
    return 0.5f * x * (1.0f + t);
}

// -------------------------------------------------------------------- feat --
__global__ __launch_bounds__(256)
void sp_feat_kernel(const float* __restrict__ feat, const float* __restrict__ pos,
                    const float* __restrict__ w_in, const float* __restrict__ b_in,
                    _Float16* __restrict__ xh) {
    int node = blockIdx.x;      // 0 .. B*N-1 (flat)
    int h    = threadIdx.x;     // 0 .. 255
    const float* f = feat + (size_t)node * IN_DIM_;
    float acc = b_in[h];
#pragma unroll
    for (int k = 0; k < IN_DIM_; ++k) acc += f[k] * w_in[k * HID_ + h];
    // ContinuousSincosEmbed: per dim d: [sin(p_d*freq_0..41) | cos(...)], pad 252->256
    if (h < 252) {
        int d = h / 84, r = h % 84, j = (r < 42) ? r : (r - 42);
        float fr  = __expf(-(float)j * (9.210340371976184f / 41.0f)); // ln(1e4)/41
        float ang = pos[(size_t)node * NDIM_ + d] * fr;
        acc += (r < 42) ? __sinf(ang) : __cosf(ang);
    }
    xh[(size_t)node * HID_ + h] = (_Float16)acc;
}

// ----------------------------------------------------------- weight convert --
__global__ __launch_bounds__(256)
void sp_wcvt_kernel(const float* __restrict__ w_m1, const float* __restrict__ w_m2,
                    _Float16* __restrict__ w1h, _Float16* __restrict__ w2h) {
    int i = blockIdx.x * blockDim.x + threadIdx.x;
    if (i < HID_ * HID_) {
        w1h[i] = (_Float16)w_m1[i];   // rows 0..255 of (512,256) = src half
        w2h[i] = (_Float16)w_m2[i];
    }
}

// -------------------------------------------------------------- edge build --
__global__ __launch_bounds__(256)
void sp_edge_kernel(const float* __restrict__ pos,
                    int* __restrict__ srcs, int* __restrict__ cnts) {
    int bs = blockIdx.x;                 // b*MAX_SUP + s
    int b  = bs >> 10;
    int s  = bs & (MAX_SUP_ - 1);
    __shared__ int cnt;
    if (threadIdx.x == 0) cnt = 0;
    __syncthreads();
    const float* pb = pos + (size_t)b * N_ * NDIM_;
    float sx = pb[s * 3 + 0], sy = pb[s * 3 + 1], sz = pb[s * 3 + 2];
    for (int n = threadIdx.x; n < N_; n += blockDim.x) {
        float dx = pb[n * 3 + 0] - sx;
        float dy = pb[n * 3 + 1] - sy;
        float dz = pb[n * 3 + 2] - sz;
        if (dx * dx + dy * dy + dz * dz <= R2_) {
            int idx = atomicAdd(&cnt, 1);
            if (idx < CAP_) srcs[(size_t)bs * CAP_ + idx] = n;
        }
    }
    __syncthreads();
    if (threadIdx.x == 0) cnts[bs] = (cnt < CAP_) ? cnt : CAP_;
}

// --------------------------------------------------- per-supernode dst term --
__global__ __launch_bounds__(256)
void sp_dvec_kernel(const _Float16* __restrict__ xh, const float* __restrict__ w_m1,
                    const float* __restrict__ b_m1, float* __restrict__ dvec) {
    int bs = blockIdx.x;
    int b  = bs >> 10;
    int s  = bs & (MAX_SUP_ - 1);
    int n  = threadIdx.x;
    const _Float16* xv = xh + ((size_t)(b * N_ + s)) * HID_;
    float acc = b_m1[n];
#pragma unroll 4
    for (int k = 0; k < HID_; ++k)
        acc += (float)xv[k] * w_m1[(size_t)(HID_ + k) * HID_ + n];
    dvec[(size_t)bs * HID_ + n] = acc;
}

// -------------------------------------------------- fused edge MLP + pool ---
__global__ __launch_bounds__(256)
void sp_mlp_pool_kernel(const _Float16* __restrict__ xh,
                        const _Float16* __restrict__ w1h,
                        const _Float16* __restrict__ w2h,
                        const float* __restrict__ dvec,
                        const float* __restrict__ b_m2,
                        const int* __restrict__ srcs,
                        const int* __restrict__ cnts,
                        float* __restrict__ out) {
    int bs = blockIdx.x;                 // b*MAX_SUP + s
    int b  = bs >> 10;

    __shared__ float accv[HID_];
    __shared__ float dv[HID_];
    __shared__ __attribute__((aligned(16))) _Float16 atile[16][HID_];
    __shared__ __attribute__((aligned(16))) _Float16 htile[16][HID_];

    int t = threadIdx.x;
    accv[t] = 0.0f;
    dv[t]   = dvec[(size_t)bs * HID_ + t];
    int cnt = cnts[bs];
    __syncthreads();

    int lane  = t & 31;
    int wave  = t >> 5;                  // 8 wave32s
    int row16 = lane & 15;
    int lh    = lane >> 4;               // lane-half: 0 or 1

    int groups = (cnt + 15) >> 4;
    for (int g = 0; g < groups; ++g) {
        int rows_valid = cnt - g * 16;
        if (rows_valid > 16) rows_valid = 16;

        // ---- stage 16 src feature rows (f16) into LDS: 256 thr x 32B ----
        {
            int arow = t >> 4, seg = t & 15;
            _Float16* dst = &atile[arow][seg * 16];
            if (arow < rows_valid) {
                int src = srcs[(size_t)bs * CAP_ + g * 16 + arow];
                const _Float16* sp = xh + ((size_t)(b * N_ + src)) * HID_ + seg * 16;
                *(v8h*)dst       = *(const v8h*)sp;
                *(v8h*)(dst + 8) = *(const v8h*)(sp + 8);
            } else {
                v8h z = {};
                *(v8h*)dst = z; *(v8h*)(dst + 8) = z;
            }
        }
        __syncthreads();

        // ---- layer 1: h1 = gelu(Xsrc @ W1_top + dvec) ; 2 col-blocks/wave --
        for (int q = 0; q < 2; ++q) {
            int nb = wave * 2 + q;                     // 16-col block index
            float dval = dv[nb * 16 + row16];          // C col = lane&15
            v8f acc;
#pragma unroll
            for (int i = 0; i < 8; ++i) acc[i] = dval;
#pragma unroll
            for (int kb = 0; kb < 8; ++kb) {
                // A frag: row = lane&15 ; K = kb*32 + 8*lh + {0..7}, {16..23}
                const _Float16* ap = &atile[row16][kb * 32 + 8 * lh];
                v8h alo = *(const v8h*)ap;
                v8h ahi = *(const v8h*)(ap + 16);
                v16h a;
#pragma unroll
                for (int i = 0; i < 8; ++i) { a[i] = alo[i]; a[i + 8] = ahi[i]; }
                // B frag: col = nb*16 + (lane&15) ; K = kb*32 + 8*lh + {0..7}, {16..23}
                const _Float16* bp =
                    w1h + (size_t)(kb * 32 + 8 * lh) * HID_ + nb * 16 + row16;
                v16h bb;
#pragma unroll
                for (int i = 0; i < 8; ++i) { bb[i] = bp[i * HID_]; bb[i + 8] = bp[(i + 16) * HID_]; }
                acc = __builtin_amdgcn_wmma_f32_16x16x32_f16(
                    false, a, false, bb, (short)0, acc, false, false);
            }
            // gelu + invalid-edge mask, spill h1 tile to LDS as f16
#pragma unroll
            for (int i = 0; i < 8; ++i) {
                int m = i + 8 * lh;                    // C/D row
                float v = (m < rows_valid) ? gelu_tanh(acc[i]) : 0.0f;
                htile[m][nb * 16 + row16] = (_Float16)v;
            }
        }
        __syncthreads();

        // ---- layer 2: o = h1 @ W2 ; row-sum valid edges into accv ----------
        for (int q = 0; q < 2; ++q) {
            int nb = wave * 2 + q;
            v8f acc = {};
#pragma unroll
            for (int kb = 0; kb < 8; ++kb) {
                const _Float16* ap = &htile[row16][kb * 32 + 8 * lh];
                v8h alo = *(const v8h*)ap;
                v8h ahi = *(const v8h*)(ap + 16);
                v16h a;
#pragma unroll
                for (int i = 0; i < 8; ++i) { a[i] = alo[i]; a[i + 8] = ahi[i]; }
                const _Float16* bp =
                    w2h + (size_t)(kb * 32 + 8 * lh) * HID_ + nb * 16 + row16;
                v16h bb;
#pragma unroll
                for (int i = 0; i < 8; ++i) { bb[i] = bp[i * HID_]; bb[i + 8] = bp[(i + 16) * HID_]; }
                acc = __builtin_amdgcn_wmma_f32_16x16x32_f16(
                    false, a, false, bb, (short)0, acc, false, false);
            }
            float partial = 0.0f;
#pragma unroll
            for (int i = 0; i < 8; ++i) {
                int m = i + 8 * lh;
                partial += (m < rows_valid) ? acc[i] : 0.0f;
            }
            atomicAdd(&accv[nb * 16 + row16], partial);   // ds f32 atomic
        }
        __syncthreads();
    }

    // mean + bias (b2 contributes once iff cnt>0; all 1024 supernodes active)
    float inv = 1.0f / (float)((cnt > 0) ? cnt : 1);
    float o   = accv[t] * inv + ((cnt > 0) ? b_m2[t] : 0.0f);
    out[(size_t)bs * HID_ + t] = o;
}

// ------------------------------------------------------------------ launch --
extern "C" void kernel_launch(void* const* d_in, const int* in_sizes, int n_in,
                              void* d_out, int out_size, void* d_ws, size_t ws_size,
                              hipStream_t stream) {
    (void)in_sizes; (void)n_in; (void)out_size; (void)ws_size;
    const float* input_feat = (const float*)d_in[0];   // (B,N,16)
    const float* input_pos  = (const float*)d_in[1];   // (B,N,3)
    // d_in[2] supernode_mask: arange(N)<MAX_SUP for every batch -> implicit
    const float* w_in = (const float*)d_in[3];         // (16,256)
    const float* b_in = (const float*)d_in[4];         // (256,)
    const float* w_m1 = (const float*)d_in[5];         // (512,256)
    const float* b_m1 = (const float*)d_in[6];         // (256,)
    const float* w_m2 = (const float*)d_in[7];         // (256,256)
    const float* b_m2 = (const float*)d_in[8];         // (256,)
    float* out = (float*)d_out;                        // (B,1024,256)

    char* ws = (char*)d_ws;
    _Float16* xh   = (_Float16*)(ws + 0);              // 4,194,304 B
    _Float16* w1h  = (_Float16*)(ws + 4194304);        //   131,072 B
    _Float16* w2h  = (_Float16*)(ws + 4325376);        //   131,072 B
    float*    dvec = (float*)   (ws + 4456448);        // 2,097,152 B
    int*      srcs = (int*)     (ws + 6553600);        //   524,288 B
    int*      cnts = (int*)     (ws + 7077888);        //     8,192 B

    sp_feat_kernel<<<B_ * N_, 256, 0, stream>>>(input_feat, input_pos, w_in, b_in, xh);
    sp_wcvt_kernel<<<(HID_ * HID_ + 255) / 256, 256, 0, stream>>>(w_m1, w_m2, w1h, w2h);
    sp_edge_kernel<<<B_ * MAX_SUP_, 256, 0, stream>>>(input_pos, srcs, cnts);
    sp_dvec_kernel<<<B_ * MAX_SUP_, 256, 0, stream>>>(xh, w_m1, b_m1, dvec);
    sp_mlp_pool_kernel<<<B_ * MAX_SUP_, 256, 0, stream>>>(xh, w1h, w2h, dvec, b_m2,
                                                          srcs, cnts, out);
}
